// GNNModel_11209864642618
// MI455X (gfx1250) — matrix operand order repaired and checked by reading, not compiled, not run on hardware
//
#include <hip/hip_runtime.h>
#include <hip/hip_bf16.h>

typedef __attribute__((ext_vector_type(16))) _Float16 v16h;
typedef __attribute__((ext_vector_type(8)))  _Float16 v8h;
typedef __attribute__((ext_vector_type(8)))  float    v8f;

#define NND 100000
#define NE  3200000
#define FIN 128
#define FHID 256
#define FOUT 64

// ---------- utility kernels ----------

__global__ void zero_f32(float* __restrict__ p, int n) {
  int i = blockIdx.x * 256 + threadIdx.x;
  if (i < n) p[i] = 0.0f;
}

__global__ void deg_kernel(const int* __restrict__ src, const int* __restrict__ dst,
                           float* __restrict__ outdeg, float* __restrict__ indeg, int E) {
  int e = blockIdx.x * 256 + threadIdx.x;
  if (e < E) {
    atomicAdd(&outdeg[src[e]], 1.0f);
    atomicAdd(&indeg[dst[e]], 1.0f);
  }
}

__global__ void invsqrt_kernel(float* __restrict__ d, int n) {
  int i = blockIdx.x * 256 + threadIdx.x;
  if (i < n) d[i] = rsqrtf(fmaxf(d[i], 1.0f));
}

// xs[n, 0:64] = lvl[n]*invout[n] ; xs[n, 64:128] = pri[n]*invout[n]
__global__ void scale_concat(const float* __restrict__ lvl, const float* __restrict__ pri,
                             const float* __restrict__ invout, float* __restrict__ xs,
                             int total) {
  int i = blockIdx.x * 256 + threadIdx.x;
  if (i >= total) return;
  int n = i >> 7, f = i & 127;
  float v = (f < 64) ? lvl[n * 64 + f] : pri[n * 64 + (f - 64)];
  xs[i] = v * invout[n];
}

// agg[dst[e], :] += xs[src[e], :]   (F = 128, 32 lanes x float4 per edge)
__global__ void edge_scatter128(const int* __restrict__ src, const int* __restrict__ dst,
                                const float* __restrict__ xs, float* __restrict__ agg, int E) {
  int t = blockIdx.x * 256 + threadIdx.x;
  int e = t >> 5;
  if (e >= E) return;
  int c = (t & 31) << 2;
  const float4 v = *(const float4*)(xs + (size_t)src[e] * 128 + c);
  float* o = agg + (size_t)dst[e] * 128 + c;
  atomicAdd(o + 0, v.x); atomicAdd(o + 1, v.y);
  atomicAdd(o + 2, v.z); atomicAdd(o + 3, v.w);
}

// agg[dst[e], :] += y[src[e], :]    (F = 64, 16 lanes x float4 per edge)
__global__ void edge_scatter64(const int* __restrict__ src, const int* __restrict__ dst,
                               const float* __restrict__ y, float* __restrict__ agg, int E) {
  int t = blockIdx.x * 256 + threadIdx.x;
  int e = t >> 4;
  if (e >= E) return;
  int c = (t & 15) << 2;
  const float4 v = *(const float4*)(y + (size_t)src[e] * 64 + c);
  float* o = agg + (size_t)dst[e] * 64 + c;
  atomicAdd(o + 0, v.x); atomicAdd(o + 1, v.y);
  atomicAdd(o + 2, v.z); atomicAdd(o + 3, v.w);
}

// a1h[n,k] = f16(agg1[n,k] * invin[n])
__global__ void cvt_scale_f16(const float* __restrict__ a, const float* __restrict__ invin,
                              _Float16* __restrict__ o, int total) {
  int i = blockIdx.x * 256 + threadIdx.x;
  if (i < total) o[i] = (_Float16)(a[i] * invin[i >> 7]);
}

// Wt[n,k] = f16(W[k,n])   (W is [K, Nc] row-major fp32)
__global__ void wt_cvt(const float* __restrict__ W, _Float16* __restrict__ Wt, int K, int Nc) {
  int i = blockIdx.x * 256 + threadIdx.x;
  if (i >= K * Nc) return;
  int k = i / Nc, n = i % Nc;
  Wt[n * K + k] = (_Float16)W[i];
}

// out[n,c] = agg2[n,c] * invin[n] + b2[c]
__global__ void final_kernel(const float* __restrict__ agg, const float* __restrict__ invin,
                             const float* __restrict__ b2, float* __restrict__ out, int total) {
  int i = blockIdx.x * 256 + threadIdx.x;
  if (i < total) {
    int n = i >> 6, c = i & 63;
    out[i] = agg[i] * invin[n] + b2[c];
  }
}

// ---------- WMMA GEMM kernels (one wave per 16x16 tile) ----------
// A-fragment (16-bit A 16x32): lane l holds row M=l&15; element i holds
//   K = kk + (l>>4)*16 + i                  -> two contiguous v8h loads.
// B-fragment (16-bit B 32x16): lane l holds column N=l&15; element j holds
//   K = kk + (l>>4)*16 + j                  -> one contiguous v16h load from Bt[N,K].
// C/D (f32 16x16): VGPR r -> M = r + (lane>>4)*8, N = lane&15.

// hs[m, 0:256] = f16( relu(A@W1 + b1) * rowscale[m] )
__global__ __launch_bounds__(256) void gemm1_wmma(
    const _Float16* __restrict__ A,        // [NND, 128] f16
    const _Float16* __restrict__ Bt,       // [256, 128] f16 (W1 transposed)
    const float* __restrict__ bias,        // [256]
    const float* __restrict__ rowscale,    // [NND] (= inv_sqrt_out, folded for layer 2)
    _Float16* __restrict__ H,              // [NND, 256] f16
    int tilesM) {
  const int lane = threadIdx.x & 31;
  const int wave = blockIdx.x * (blockDim.x >> 5) + (threadIdx.x >> 5);
  const int tilesN = FHID / 16;
  const int tileM = wave / tilesN;
  const int tileN = wave % tilesN;
  if (tileM >= tilesM) return;

  const int arow = tileM * 16 + (lane & 15);
  const int ka   = (lane >> 4) * 16;
  const int bn   = tileN * 16 + (lane & 15);
  const int kb   = (lane >> 4) * 16;
  const _Float16* Arow = A  + (size_t)arow * FIN;
  const _Float16* Bcol = Bt + (size_t)bn * FIN;

  v8f c = {};
#pragma unroll
  for (int kk = 0; kk < FIN; kk += 32) {
    v8h alo = *(const v8h*)(Arow + kk + ka);
    v8h ahi = *(const v8h*)(Arow + kk + ka + 8);
    v16h a;
#pragma unroll
    for (int i = 0; i < 8; ++i) { a[i] = alo[i]; a[i + 8] = ahi[i]; }
    v16h b = *(const v16h*)(Bcol + kk + kb);
    c = __builtin_amdgcn_wmma_f32_16x16x32_f16(false, a, false, b, (short)0, c, false, false);
  }

  const int mbase = tileM * 16 + (lane >> 4) * 8;
  const int n = tileN * 16 + (lane & 15);
  const float bi = bias[n];
#pragma unroll
  for (int r = 0; r < 8; ++r) {
    int m = mbase + r;
    float v = fmaxf(c[r] + bi, 0.0f) * rowscale[m];
    H[(size_t)m * FHID + n] = (_Float16)v;
  }
}

// y[m, 0:64] = fp32( H @ W2 )   (bias/norm applied after aggregation)
__global__ __launch_bounds__(256) void gemm2_wmma(
    const _Float16* __restrict__ A,        // [NND, 256] f16
    const _Float16* __restrict__ Bt,       // [64, 256] f16 (W2 transposed)
    float* __restrict__ Y,                 // [NND, 64] fp32
    int tilesM) {
  const int lane = threadIdx.x & 31;
  const int wave = blockIdx.x * (blockDim.x >> 5) + (threadIdx.x >> 5);
  const int tilesN = FOUT / 16;
  const int tileM = wave / tilesN;
  const int tileN = wave % tilesN;
  if (tileM >= tilesM) return;

  const int arow = tileM * 16 + (lane & 15);
  const int ka   = (lane >> 4) * 16;
  const int bn   = tileN * 16 + (lane & 15);
  const int kb   = (lane >> 4) * 16;
  const _Float16* Arow = A  + (size_t)arow * FHID;
  const _Float16* Bcol = Bt + (size_t)bn * FHID;

  v8f c = {};
#pragma unroll
  for (int kk = 0; kk < FHID; kk += 32) {
    v8h alo = *(const v8h*)(Arow + kk + ka);
    v8h ahi = *(const v8h*)(Arow + kk + ka + 8);
    v16h a;
#pragma unroll
    for (int i = 0; i < 8; ++i) { a[i] = alo[i]; a[i + 8] = ahi[i]; }
    v16h b = *(const v16h*)(Bcol + kk + kb);
    c = __builtin_amdgcn_wmma_f32_16x16x32_f16(false, a, false, b, (short)0, c, false, false);
  }

  const int mbase = tileM * 16 + (lane >> 4) * 8;
  const int n = tileN * 16 + (lane & 15);
#pragma unroll
  for (int r = 0; r < 8; ++r) {
    Y[(size_t)(mbase + r) * FOUT + n] = c[r];
  }
}

// ---------- launch ----------

extern "C" void kernel_launch(void* const* d_in, const int* in_sizes, int n_in,
                              void* d_out, int out_size, void* d_ws, size_t ws_size,
                              hipStream_t stream) {
  (void)in_sizes; (void)n_in; (void)out_size; (void)ws_size;
  const float* lvl = (const float*)d_in[0];
  const float* pri = (const float*)d_in[1];
  const int*   src = (const int*)d_in[2];
  const int*   dst = (const int*)d_in[3];
  const float* W1  = (const float*)d_in[4];
  const float* b1  = (const float*)d_in[5];
  const float* W2  = (const float*)d_in[6];
  const float* b2  = (const float*)d_in[7];
  float* out = (float*)d_out;

  char* ws = (char*)d_ws;
  size_t off = 0;
  auto alloc = [&](size_t bytes) -> char* {
    char* p = ws + off;
    off += (bytes + 255) & ~(size_t)255;
    return p;
  };

  float*    invout = (float*)alloc((size_t)NND * 4);
  float*    invin  = (float*)alloc((size_t)NND * 4);
  char*     R1     = alloc((size_t)NND * FIN * 4);   // xs -> later y | agg2
  char*     R2     = alloc((size_t)NND * FIN * 4);   // agg1 -> later hs (f16 [NND,256])
  _Float16* a1h    = (_Float16*)alloc((size_t)NND * FIN * 2);
  _Float16* w1t    = (_Float16*)alloc((size_t)FIN * FHID * 2);
  _Float16* w2t    = (_Float16*)alloc((size_t)FHID * FOUT * 2);

  float*    xs   = (float*)R1;
  float*    agg1 = (float*)R2;
  _Float16* hs   = (_Float16*)R2;                    // overlays agg1 (same byte size)
  float*    y    = (float*)R1;                       // overlays dead xs
  float*    agg2 = (float*)(R1 + (size_t)NND * FOUT * 4);

  const int B = 256;
  auto nb = [](long long n) { return (int)((n + 255) / 256); };

  // degrees -> inverse sqrt norms
  zero_f32<<<nb(NND), B, 0, stream>>>(invout, NND);
  zero_f32<<<nb(NND), B, 0, stream>>>(invin, NND);
  deg_kernel<<<nb(NE), B, 0, stream>>>(src, dst, invout, invin, NE);
  invsqrt_kernel<<<nb(NND), B, 0, stream>>>(invout, NND);
  invsqrt_kernel<<<nb(NND), B, 0, stream>>>(invin, NND);

  // layer 1: xs = concat(lvl,pri) * invout ; agg1 = scatter-add over edges
  scale_concat<<<nb((long long)NND * FIN), B, 0, stream>>>(lvl, pri, invout, xs, NND * FIN);
  zero_f32<<<nb((long long)NND * FIN), B, 0, stream>>>(agg1, NND * FIN);
  edge_scatter128<<<nb((long long)NE * 32), B, 0, stream>>>(src, dst, xs, agg1, NE);

  // a1h = f16(agg1 * invin) ; weights -> transposed f16
  cvt_scale_f16<<<nb((long long)NND * FIN), B, 0, stream>>>(agg1, invin, a1h, NND * FIN);
  wt_cvt<<<nb((long long)FIN * FHID), B, 0, stream>>>(W1, w1t, FIN, FHID);
  wt_cvt<<<nb((long long)FHID * FOUT), B, 0, stream>>>(W2, w2t, FHID, FOUT);

  // GEMM1 (WMMA f16->f32): hs = f16(relu(a1h @ W1 + b1) * invout)
  {
    int tilesM = NND / 16;                 // 6250
    int waves  = tilesM * (FHID / 16);     // 100000
    gemm1_wmma<<<waves / 8, B, 0, stream>>>(a1h, w1t, b1, invout, hs, tilesM);
  }

  // GEMM2 (WMMA f16->f32): y = hs @ W2   (invout already folded into hs)
  {
    int tilesM = NND / 16;
    int waves  = tilesM * (FOUT / 16);     // 25000
    gemm2_wmma<<<waves / 8, B, 0, stream>>>(hs, w2t, y, tilesM);
  }

  // layer 2 aggregation over 64 features, then final norm + bias
  zero_f32<<<nb((long long)NND * FOUT), B, 0, stream>>>(agg2, NND * FOUT);
  edge_scatter64<<<nb((long long)NE * 16), B, 0, stream>>>(src, dst, y, agg2, NE);
  final_kernel<<<nb((long long)NND * FOUT), B, 0, stream>>>(agg2, invin, b2, out, NND * FOUT);
}
